// Attention_84232898609328
// MI455X (gfx1250) — matrix operand order repaired and checked by reading, not compiled, not run on hardware
//
#include <hip/hip_runtime.h>

// ---------------------------------------------------------------------------
// CDNA5 (gfx1250, wave32) attention with L2-normalized scores.
// All matmuls on v_wmma_f32_16x16x32_bf16 (f32 accumulate).
// Tile staging uses GLOBAL_LOAD_ASYNC_TO_LDS_B128 (ASYNCcnt) where no data
// permutation is needed; V-transpose keeps the VGPR bounce path.
// Two-pass attention: pass1 accumulates row sum-of-squares of QK^T without
// materializing the 512MB score matrix; pass2 recomputes QK^T tiles, scales,
// and multiplies by V. Workspace use: ~48.3 MB.
// ---------------------------------------------------------------------------

typedef __bf16 bf16;
typedef __attribute__((ext_vector_type(16))) __bf16 v16bf;
typedef __attribute__((ext_vector_type(8)))  float  v8f;

#define DIM   1024
#define HEADS 16
#define HD    64
#define SEQ   2048
#define BATCH 2
#define ROWS  (BATCH * SEQ)   // 4096

__device__ __forceinline__ bf16 f2bf(float f) {
  unsigned u = __float_as_uint(f);
  u += 0x7FFFu + ((u >> 16) & 1u);          // round-to-nearest-even
  unsigned short h = (unsigned short)(u >> 16);
  return __builtin_bit_cast(bf16, h);
}

union Frag32B { uint4 q[2]; v16bf v; };

// A fragment (16x32 bf16, ISA 7.12.2): lane=16*hi+m holds K[8hi..8hi+7] ++ K[16+8hi..+7]
__device__ __forceinline__ v16bf frag_a(const bf16* sm, int stride, int row, int hi) {
  Frag32B f;
  const bf16* p = sm + row * stride + hi * 8;
  f.q[0] = *(const uint4*)(p);
  f.q[1] = *(const uint4*)(p + 16);
  return f.v;
}
// B fragment (32x16 bf16): lane=16*hi+n holds K[8hi..8hi+7] ++ K[16+8hi..+7] of column n
__device__ __forceinline__ v16bf frag_b(const bf16* sm, int stride, int col, int hi) {
  Frag32B f;
  const bf16* p = sm + col * stride + hi * 8;
  f.q[0] = *(const uint4*)(p);
  f.q[1] = *(const uint4*)(p + 16);
  return f.v;
}

__device__ __forceinline__ v8f wmma_bf16(v16bf a, v16bf b, v8f c) {
  return __builtin_amdgcn_wmma_f32_16x16x32_bf16(false, a, false, b, (short)0, c,
                                                 false, false);
}

// Async global->LDS 16B copy (per-lane), tracked with ASYNCcnt.
// Low 32 bits of a generic pointer to __shared__ are the LDS byte offset,
// matching GLOBAL_LOAD_ASYNC_TO_LDS's VDST semantics.
__device__ __forceinline__ void async_b128(const bf16* g, bf16* l) {
  unsigned lo = (unsigned)(uintptr_t)l;
  asm volatile("global_load_async_to_lds_b128 %0, %1, off"
               :: "v"(lo), "v"(g) : "memory");
}
__device__ __forceinline__ void wait_async0() {
  asm volatile("s_wait_asynccnt 0x0" ::: "memory");
}

// ---------------------------------------------------------------------------
// fp32 -> bf16 conversion (one-time cost so GEMMs stream pure bf16)
// ---------------------------------------------------------------------------
__global__ __launch_bounds__(256) void k_cvt(const float* __restrict__ in,
                                             bf16* __restrict__ out, int n) {
  int i = blockIdx.x * 256 + threadIdx.x;
  if (i < n) out[i] = f2bf(in[i]);
}

// ---------------------------------------------------------------------------
// QKV GEMM: [4096,1024] x [3072,1024]^T, scatter into Q/K/V [b][h][n][d] bf16
// ---------------------------------------------------------------------------
__global__ __launch_bounds__(256) void k_gemm_qkv(const bf16* __restrict__ xb,
                                                  const bf16* __restrict__ wqb,
                                                  bf16* __restrict__ Qb,
                                                  bf16* __restrict__ Kb,
                                                  bf16* __restrict__ Vb) {
  __shared__ bf16 smA[128 * 40];
  __shared__ bf16 smB[128 * 40];
  const int tid = threadIdx.x, lane = tid & 31, hi = lane >> 4, l15 = lane & 15;
  const int w = tid >> 5, wr = w & 3, wc = w >> 2;
  const int m0 = blockIdx.y * 128, n0 = blockIdx.x * 128;

  const v8f z = {0.f, 0.f, 0.f, 0.f, 0.f, 0.f, 0.f, 0.f};
  v8f acc[2][4];
#pragma unroll
  for (int mi = 0; mi < 2; ++mi)
#pragma unroll
    for (int ni = 0; ni < 4; ++ni) acc[mi][ni] = z;

  for (int kt = 0; kt < DIM; kt += 32) {
#pragma unroll
    for (int r = 0; r < 2; ++r) {
      int i = tid + r * 256, row = i >> 2, seg = i & 3;
      async_b128(xb + (size_t)(m0 + row) * DIM + kt + seg * 8,
                 smA + row * 40 + seg * 8);
      async_b128(wqb + (size_t)(n0 + row) * DIM + kt + seg * 8,
                 smB + row * 40 + seg * 8);
    }
    if (kt + 32 < DIM) {
      __builtin_prefetch(xb + (size_t)(m0 + (tid >> 1)) * DIM + kt + 32, 0, 1);
      __builtin_prefetch(wqb + (size_t)(n0 + (tid >> 1)) * DIM + kt + 32, 0, 1);
    }
    wait_async0();
    __syncthreads();
    v16bf af[2], bfv[4];
#pragma unroll
    for (int mi = 0; mi < 2; ++mi) af[mi] = frag_a(smA, 40, wr * 32 + mi * 16 + l15, hi);
#pragma unroll
    for (int ni = 0; ni < 4; ++ni) bfv[ni] = frag_b(smB, 40, wc * 64 + ni * 16 + l15, hi);
#pragma unroll
    for (int mi = 0; mi < 2; ++mi)
#pragma unroll
      for (int ni = 0; ni < 4; ++ni) acc[mi][ni] = wmma_bf16(af[mi], bfv[ni], acc[mi][ni]);
    __syncthreads();
  }

#pragma unroll
  for (int mi = 0; mi < 2; ++mi) {
    int gm = m0 + wr * 32 + mi * 16 + 8 * hi;
#pragma unroll
    for (int ni = 0; ni < 4; ++ni) {
      int o = n0 + wc * 64 + ni * 16 + l15;
      int s = o >> 10, hh = (o >> 6) & 15, dd = o & 63;
      bf16* dst = (s == 0) ? Qb : ((s == 1) ? Kb : Vb);
#pragma unroll
      for (int r = 0; r < 8; ++r) {
        int row = gm + r, b = row >> 11, n = row & 2047;
        dst[(((size_t)b * HEADS + hh) * SEQ + n) * HD + dd] = f2bf(acc[mi][ni][r]);
      }
    }
  }
}

// ---------------------------------------------------------------------------
// Pass 1: per-row inverse L2 norm of QK^T (scores never hit HBM)
// ---------------------------------------------------------------------------
__global__ __launch_bounds__(256) void k_rownorm(const bf16* __restrict__ Qb,
                                                 const bf16* __restrict__ Kb,
                                                 float* __restrict__ rnorm, float eps) {
  __shared__ bf16 smQ[128 * 72];
  __shared__ bf16 smK[128 * 72];
  __shared__ float smRS[128];
  const int tid = threadIdx.x, lane = tid & 31, hi = lane >> 4, l15 = lane & 15;
  const int w = tid >> 5, wr = w & 3, wc = w >> 2;
  const int bh = blockIdx.x >> 4;
  const int i0 = (blockIdx.x & 15) * 128;
  const bf16* Qp = Qb + ((size_t)bh * SEQ + i0) * HD;
  const bf16* Kp = Kb + (size_t)bh * SEQ * HD;

#pragma unroll
  for (int r = 0; r < 4; ++r) {
    int i = tid + r * 256, row = i >> 3, seg = i & 7;
    async_b128(Qp + (size_t)row * HD + seg * 8, smQ + row * 72 + seg * 8);
  }
  if (tid < 128) smRS[tid] = 0.f;

  float ssAcc[2][8];
#pragma unroll
  for (int mi = 0; mi < 2; ++mi)
#pragma unroll
    for (int r = 0; r < 8; ++r) ssAcc[mi][r] = 0.f;

  const v8f z = {0.f, 0.f, 0.f, 0.f, 0.f, 0.f, 0.f, 0.f};
  for (int jt = 0; jt < 16; ++jt) {
    __syncthreads();
#pragma unroll
    for (int r = 0; r < 4; ++r) {
      int i = tid + r * 256, row = i >> 3, seg = i & 7;
      async_b128(Kp + (size_t)(jt * 128 + row) * HD + seg * 8,
                 smK + row * 72 + seg * 8);
    }
    wait_async0();
    __syncthreads();
    v8f s[2][4];
#pragma unroll
    for (int mi = 0; mi < 2; ++mi)
#pragma unroll
      for (int ni = 0; ni < 4; ++ni) s[mi][ni] = z;
#pragma unroll
    for (int kc = 0; kc < 2; ++kc) {
      v16bf af[2], bfv[4];
#pragma unroll
      for (int mi = 0; mi < 2; ++mi)
        af[mi] = frag_a(smQ + kc * 32, 72, wr * 32 + mi * 16 + l15, hi);
#pragma unroll
      for (int ni = 0; ni < 4; ++ni)
        bfv[ni] = frag_b(smK + kc * 32, 72, wc * 64 + ni * 16 + l15, hi);
#pragma unroll
      for (int mi = 0; mi < 2; ++mi)
#pragma unroll
        for (int ni = 0; ni < 4; ++ni) s[mi][ni] = wmma_bf16(af[mi], bfv[ni], s[mi][ni]);
    }
#pragma unroll
    for (int mi = 0; mi < 2; ++mi)
#pragma unroll
      for (int r = 0; r < 8; ++r) {
        float ss = 0.f;
#pragma unroll
        for (int ni = 0; ni < 4; ++ni) ss += s[mi][ni][r] * s[mi][ni][r];
        ssAcc[mi][r] += ss;
      }
  }
  // cross-lane (16-lane group) reduction, then LDS atomic across waves
#pragma unroll
  for (int mi = 0; mi < 2; ++mi)
#pragma unroll
    for (int r = 0; r < 8; ++r) {
      float ss = ssAcc[mi][r];
      ss += __shfl_xor(ss, 1, 16);
      ss += __shfl_xor(ss, 2, 16);
      ss += __shfl_xor(ss, 4, 16);
      ss += __shfl_xor(ss, 8, 16);
      if (l15 == 0) atomicAdd(&smRS[wr * 32 + mi * 16 + r + 8 * hi], ss);
    }
  __syncthreads();
  if (tid < 128) {
    float nrm = sqrtf(smRS[tid]);
    rnorm[(size_t)bh * SEQ + i0 + tid] = 1.f / fmaxf(nrm, eps);
  }
}

// ---------------------------------------------------------------------------
// Pass 2: recompute QK^T tile, scale (rnorm*g + b), multiply by V -> ctx bf16
// ---------------------------------------------------------------------------
__global__ __launch_bounds__(256) void k_attn_av(const bf16* __restrict__ Qb,
                                                 const bf16* __restrict__ Kb,
                                                 const bf16* __restrict__ Vb,
                                                 const float* __restrict__ rnorm,
                                                 const float* __restrict__ gvec,
                                                 const float* __restrict__ bvec,
                                                 bf16* __restrict__ ctx) {
  __shared__ bf16 smQ[128 * 72];
  __shared__ bf16 smK[64 * 72];
  __shared__ bf16 smVt[64 * 72];     // V transposed: [d][j]
  __shared__ bf16 smAttn[128 * 72];  // attn tile bf16: [i][j]
  __shared__ float smRn[128];
  const int tid = threadIdx.x, lane = tid & 31, hi = lane >> 4, l15 = lane & 15;
  const int w = tid >> 5, wr = w & 3, wc = w >> 2;
  const int bh = blockIdx.x >> 4, h = bh & 15, bb = bh >> 4;
  const int i0 = (blockIdx.x & 15) * 128;
  const bf16* Qp = Qb + ((size_t)bh * SEQ + i0) * HD;
  const bf16* Kp = Kb + (size_t)bh * SEQ * HD;
  const bf16* Vp = Vb + (size_t)bh * SEQ * HD;
  const float nb = bvec[h];

#pragma unroll
  for (int r = 0; r < 4; ++r) {
    int i = tid + r * 256, row = i >> 3, seg = i & 7;
    async_b128(Qp + (size_t)row * HD + seg * 8, smQ + row * 72 + seg * 8);
  }
  if (tid < 128) smRn[tid] = rnorm[(size_t)bh * SEQ + i0 + tid] * gvec[h];

  const v8f z = {0.f, 0.f, 0.f, 0.f, 0.f, 0.f, 0.f, 0.f};
  v8f oacc[2][2];
#pragma unroll
  for (int mi = 0; mi < 2; ++mi)
#pragma unroll
    for (int ni = 0; ni < 2; ++ni) oacc[mi][ni] = z;

  for (int jt = 0; jt < 32; ++jt) {
    __syncthreads();
#pragma unroll
    for (int r = 0; r < 2; ++r) {  // K tile 64x64 (async, no permutation)
      int i = tid + r * 256, row = i >> 3, seg = i & 7;
      async_b128(Kp + (size_t)(jt * 64 + row) * HD + seg * 8,
                 smK + row * 72 + seg * 8);
    }
#pragma unroll
    for (int t = 0; t < 8; ++t) {  // V tile 64x64, transposed into LDS
      int i = tid + t * 256, j = i >> 5, dp = (i & 31) * 2;
      ushort2 v = *(const ushort2*)(Vp + (size_t)(jt * 64 + j) * HD + dp);
      *(unsigned short*)(smVt + dp * 72 + j) = v.x;
      *(unsigned short*)(smVt + (dp + 1) * 72 + j) = v.y;
    }
    wait_async0();
    __syncthreads();
    // S = Q K^T  (128 x 64)
    v8f s[2][2];
#pragma unroll
    for (int mi = 0; mi < 2; ++mi)
#pragma unroll
      for (int ni = 0; ni < 2; ++ni) s[mi][ni] = z;
#pragma unroll
    for (int kc = 0; kc < 2; ++kc) {
      v16bf af[2], bfv[2];
#pragma unroll
      for (int mi = 0; mi < 2; ++mi)
        af[mi] = frag_a(smQ + kc * 32, 72, wr * 32 + mi * 16 + l15, hi);
#pragma unroll
      for (int ni = 0; ni < 2; ++ni)
        bfv[ni] = frag_b(smK + kc * 32, 72, wc * 32 + ni * 16 + l15, hi);
#pragma unroll
      for (int mi = 0; mi < 2; ++mi)
#pragma unroll
        for (int ni = 0; ni < 2; ++ni) s[mi][ni] = wmma_bf16(af[mi], bfv[ni], s[mi][ni]);
    }
    // scale and park attn tile in LDS as bf16
#pragma unroll
    for (int mi = 0; mi < 2; ++mi)
#pragma unroll
      for (int r = 0; r < 8; ++r) {
        int row = wr * 32 + mi * 16 + r + 8 * hi;
        float rg = smRn[row];
#pragma unroll
        for (int ni = 0; ni < 2; ++ni) {
          int col = wc * 32 + ni * 16 + l15;
          smAttn[row * 72 + col] = f2bf(s[mi][ni][r] * rg + nb);
        }
      }
    __syncthreads();
    // O += attn @ V   (K = 64 this tile)
#pragma unroll
    for (int kc = 0; kc < 2; ++kc) {
      v16bf af2[2], bv2[2];
#pragma unroll
      for (int mi = 0; mi < 2; ++mi)
        af2[mi] = frag_a(smAttn + kc * 32, 72, wr * 32 + mi * 16 + l15, hi);
#pragma unroll
      for (int ni = 0; ni < 2; ++ni)
        bv2[ni] = frag_b(smVt + kc * 32, 72, wc * 32 + ni * 16 + l15, hi);
#pragma unroll
      for (int mi = 0; mi < 2; ++mi)
#pragma unroll
        for (int ni = 0; ni < 2; ++ni)
          oacc[mi][ni] = wmma_bf16(af2[mi], bv2[ni], oacc[mi][ni]);
    }
  }
  // write ctx [b][n][dim] bf16
#pragma unroll
  for (int mi = 0; mi < 2; ++mi)
#pragma unroll
    for (int ni = 0; ni < 2; ++ni)
#pragma unroll
      for (int r = 0; r < 8; ++r) {
        int row = i0 + wr * 32 + mi * 16 + r + 8 * hi;
        int col = wc * 32 + ni * 16 + l15;
        ctx[((size_t)bb * SEQ + row) * DIM + h * HD + col] = f2bf(oacc[mi][ni][r]);
      }
}

// ---------------------------------------------------------------------------
// Output GEMM: [4096,1024] x [1024,1024]^T + bias -> f32 d_out
// ---------------------------------------------------------------------------
__global__ __launch_bounds__(256) void k_gemm_out(const bf16* __restrict__ ctx,
                                                  const bf16* __restrict__ wob,
                                                  const float* __restrict__ bias,
                                                  float* __restrict__ out) {
  __shared__ bf16 smA[128 * 40];
  __shared__ bf16 smB[128 * 40];
  const int tid = threadIdx.x, lane = tid & 31, hi = lane >> 4, l15 = lane & 15;
  const int w = tid >> 5, wr = w & 3, wc = w >> 2;
  const int m0 = blockIdx.y * 128, n0 = blockIdx.x * 128;

  const v8f z = {0.f, 0.f, 0.f, 0.f, 0.f, 0.f, 0.f, 0.f};
  v8f acc[2][4];
#pragma unroll
  for (int mi = 0; mi < 2; ++mi)
#pragma unroll
    for (int ni = 0; ni < 4; ++ni) acc[mi][ni] = z;

  for (int kt = 0; kt < DIM; kt += 32) {
#pragma unroll
    for (int r = 0; r < 2; ++r) {
      int i = tid + r * 256, row = i >> 2, seg = i & 3;
      async_b128(ctx + (size_t)(m0 + row) * DIM + kt + seg * 8,
                 smA + row * 40 + seg * 8);
      async_b128(wob + (size_t)(n0 + row) * DIM + kt + seg * 8,
                 smB + row * 40 + seg * 8);
    }
    if (kt + 32 < DIM) {
      __builtin_prefetch(ctx + (size_t)(m0 + (tid >> 1)) * DIM + kt + 32, 0, 1);
      __builtin_prefetch(wob + (size_t)(n0 + (tid >> 1)) * DIM + kt + 32, 0, 1);
    }
    wait_async0();
    __syncthreads();
    v16bf af[2], bfv[4];
#pragma unroll
    for (int mi = 0; mi < 2; ++mi) af[mi] = frag_a(smA, 40, wr * 32 + mi * 16 + l15, hi);
#pragma unroll
    for (int ni = 0; ni < 4; ++ni) bfv[ni] = frag_b(smB, 40, wc * 64 + ni * 16 + l15, hi);
#pragma unroll
    for (int mi = 0; mi < 2; ++mi)
#pragma unroll
      for (int ni = 0; ni < 4; ++ni) acc[mi][ni] = wmma_bf16(af[mi], bfv[ni], acc[mi][ni]);
    __syncthreads();
  }

#pragma unroll
  for (int mi = 0; mi < 2; ++mi) {
    int gm = m0 + wr * 32 + mi * 16 + 8 * hi;
#pragma unroll
    for (int ni = 0; ni < 4; ++ni) {
      int o = n0 + wc * 64 + ni * 16 + l15;
      float bv = bias[o];
#pragma unroll
      for (int r = 0; r < 8; ++r)
        out[(size_t)(gm + r) * DIM + o] = acc[mi][ni][r] + bv;
    }
  }
}

// ---------------------------------------------------------------------------
extern "C" void kernel_launch(void* const* d_in, const int* in_sizes, int n_in,
                              void* d_out, int out_size, void* d_ws, size_t ws_size,
                              hipStream_t stream) {
  const float* x     = (const float*)d_in[0];
  const float* w_qkv = (const float*)d_in[1];
  const float* w_out = (const float*)d_in[2];
  const float* b_out = (const float*)d_in[3];
  const float* ng    = (const float*)d_in[4];
  const float* nbv   = (const float*)d_in[5];
  float* out = (float*)d_out;

  // workspace carve-up (~48.3 MB total)
  char* p = (char*)d_ws;
  bf16* xb  = (bf16*)p; p += (size_t)ROWS * DIM * 2;
  bf16* wqb = (bf16*)p; p += (size_t)3 * DIM * DIM * 2;
  bf16* wob = (bf16*)p; p += (size_t)DIM * DIM * 2;
  bf16* Qb  = (bf16*)p; p += (size_t)ROWS * DIM * 2;
  bf16* Kb  = (bf16*)p; p += (size_t)ROWS * DIM * 2;
  bf16* Vb  = (bf16*)p; p += (size_t)ROWS * DIM * 2;
  bf16* ctx = (bf16*)p; p += (size_t)ROWS * DIM * 2;
  float* rn = (float*)p; p += (size_t)BATCH * HEADS * SEQ * 4;
  (void)in_sizes; (void)n_in; (void)out_size; (void)ws_size;

  k_cvt<<<(ROWS * DIM + 255) / 256, 256, 0, stream>>>(x, xb, ROWS * DIM);
  k_cvt<<<(3 * DIM * DIM + 255) / 256, 256, 0, stream>>>(w_qkv, wqb, 3 * DIM * DIM);
  k_cvt<<<(DIM * DIM + 255) / 256, 256, 0, stream>>>(w_out, wob, DIM * DIM);

  k_gemm_qkv<<<dim3(24, 32), 256, 0, stream>>>(xb, wqb, Qb, Kb, Vb);
  k_rownorm<<<512, 256, 0, stream>>>(Qb, Kb, rn, 1e-12f);
  k_attn_av<<<512, 256, 0, stream>>>(Qb, Kb, Vb, rn, ng, nbv, ctx);
  k_gemm_out<<<dim3(8, 32), 256, 0, stream>>>(ctx, wob, b_out, out);
}
